// DGCNN_Aux_T_65034394796479
// MI455X (gfx1250) — matrix operand order repaired and checked by reading, not compiled, not run on hardware
//
#include <hip/hip_runtime.h>
#include <math.h>

typedef __bf16 bf16_t;
typedef __bf16 v16bf __attribute__((ext_vector_type(16)));
typedef float  v8f   __attribute__((ext_vector_type(8)));

#define BN_SCALE 0.9999950000374997f

// ---------------- WMMA fragment helpers (CDNA5 16-bit layouts) ----------------
// A (16x32 bf16): row = m0 + (lane&15); lanes<16 hold K{0..7,16..23}, lanes>=16 hold K{8..15,24..31}
__device__ __forceinline__ v16bf load_a_frag(const bf16_t* base, int ld, int m0, int k0) {
  int lane = threadIdx.x & 31;
  int row  = m0 + (lane & 15);
  int kb   = k0 + ((lane & 16) ? 8 : 0);
  const bf16_t* p = base + row * ld + kb;
  v16bf a;
#pragma unroll
  for (int t = 0; t < 8; ++t) a[t] = p[t];
#pragma unroll
  for (int t = 0; t < 8; ++t) a[8 + t] = p[16 + t];
  return a;
}

// B (32x16 bf16), staged as [n][k] (i.e. W row-major): col n = n0+(lane&15);
// lanes<16 hold K{0..7,16..23}, lanes>=16 hold K{8..15,24..31} (same as A).
__device__ __forceinline__ v16bf load_b_frag(const bf16_t* base, int ld, int n0, int k0) {
  int lane = threadIdx.x & 31;
  int n  = n0 + (lane & 15);
  int kk = k0 + ((lane & 16) ? 8 : 0);
  const bf16_t* p = base + n * ld + kk;
  v16bf b;
#pragma unroll
  for (int t = 0; t < 8; ++t) b[t] = p[t];
#pragma unroll
  for (int t = 0; t < 8; ++t) b[8 + t] = p[16 + t];
  return b;
}

__device__ __forceinline__ v8f wmma_bf16(v16bf a, v16bf b, v8f c) {
  return __builtin_amdgcn_wmma_f32_16x16x32_bf16(false, a, false, b, (short)0, c, false, false);
}

__device__ __forceinline__ void atomicMaxF(float* addr, float v) {
  if (v >= 0.0f) atomicMax((int*)addr, __float_as_int(v));
  else           atomicMin((unsigned int*)addr, __float_as_uint(v));
}

// ---------------- input assembly: x0 = [pc | broadcast fs | zero pad] --------
__global__ void k_build_x0(const float* __restrict__ pc, const float* __restrict__ fs,
                           float* __restrict__ x0) {
  int gid = blockIdx.x * blockDim.x + threadIdx.x;
  if (gid >= 64 * 512) return;
  int f = gid >> 9;
  float* o = x0 + gid * 32;
  const float* p = pc + gid * 3;
  o[0] = p[0]; o[1] = p[1]; o[2] = p[2];
  const float* s = fs + f * 12;
#pragma unroll
  for (int c = 0; c < 12; ++c) o[3 + c] = s[c];
#pragma unroll
  for (int c = 15; c < 32; ++c) o[c] = 0.f;
}

// ---------------- kNN (K=30 smallest L2 dists, self included) ----------------
#define KNN 30
__global__ void __launch_bounds__(256) k_knn(const float* __restrict__ x, int* __restrict__ idx) {
  __shared__ float xs[512 * 32]; // 64 KB
  int f = blockIdx.x >> 1;
  int half = blockIdx.x & 1;
  const float* xf = x + f * 512 * 32;
  for (int e = threadIdx.x; e < 512 * 32; e += 256) xs[e] = xf[e];
  __syncthreads();
  int i = half * 256 + threadIdx.x;
  float bd[KNN]; int bi[KNN];
#pragma unroll
  for (int s = 0; s < KNN; ++s) { bd[s] = 3.4e38f; bi[s] = 0; }
  float xiq[32];
#pragma unroll
  for (int c = 0; c < 32; ++c) xiq[c] = xs[i * 32 + c];
  for (int j = 0; j < 512; ++j) {
    const float* xj = xs + j * 32;
    float d = 0.f;
#pragma unroll
    for (int c = 0; c < 32; ++c) { float t = xiq[c] - xj[c]; d = fmaf(t, t, d); }
    if (d < bd[KNN - 1]) {
      bd[KNN - 1] = d; bi[KNN - 1] = j;
#pragma unroll
      for (int s = KNN - 1; s > 0; --s) {
        if (bd[s] < bd[s - 1]) {
          float td = bd[s]; bd[s] = bd[s - 1]; bd[s - 1] = td;
          int ti = bi[s]; bi[s] = bi[s - 1]; bi[s - 1] = ti;
        }
      }
    }
  }
  int* o = idx + (f * 512 + i) * 32;
#pragma unroll
  for (int s = 0; s < KNN; ++s) o[s] = bi[s];
  o[30] = bi[KNN - 1]; o[31] = bi[KNN - 1];
}

// ---------------- fused EdgeConv: h build -> 3x WMMA GEMM -> max over K ------
__device__ __forceinline__ void load_w32(bf16_t* wl, const float* __restrict__ w, int fi) {
  for (int e = threadIdx.x; e < 32 * 64; e += 256) {
    int n = e >> 6, k = e & 63;
    wl[e] = (bf16_t)((k < fi) ? w[n * fi + k] : 0.f);
  }
}

// A: [256][lda] bf16 in LDS; W: [32][64] bf16 in LDS; out: [256][32] bf16 in LDS
__device__ __forceinline__ void edge_gemm(const bf16_t* A, int lda, int ktiles,
                                          const bf16_t* W, bf16_t* out,
                                          const float* __restrict__ bias,
                                          const float* __restrict__ g,
                                          const float* __restrict__ be) {
  int wave = threadIdx.x >> 5;
  int lane = threadIdx.x & 31;
#pragma unroll
  for (int jj = 0; jj < 4; ++jj) {
    int job = wave + jj * 8;       // 16 m-tiles x 2 n-tiles = 32 jobs
    int m0 = (job >> 1) * 16;
    int n0 = (job & 1) * 16;
    v8f acc = {0.f, 0.f, 0.f, 0.f, 0.f, 0.f, 0.f, 0.f};
    for (int kt = 0; kt < ktiles; ++kt) {
      v16bf a = load_a_frag(A, lda, m0, kt * 32);
      v16bf b = load_b_frag(W, 64, n0, kt * 32);
      acc = wmma_bf16(a, b, acc);
    }
    int col  = n0 + (lane & 15);
    int roff = (lane & 16) ? 8 : 0;
    float bb  = bias[col];
    float gg  = g  ? g[col]  : 0.f;
    float bee = be ? be[col] : 0.f;
#pragma unroll
    for (int r = 0; r < 8; ++r) {
      float v = acc[r] + bb;
      if (g) { v = fmaxf(v * BN_SCALE * gg + bee, 0.f); }
      out[(m0 + roff + r) * 32 + col] = (bf16_t)v;
    }
  }
}

__global__ void __launch_bounds__(256) k_edge_conv(
    const float* __restrict__ x, const int* __restrict__ idx,
    const float* __restrict__ w1, const float* __restrict__ b1,
    const float* __restrict__ g1, const float* __restrict__ be1,
    const float* __restrict__ w2, const float* __restrict__ b2,
    const float* __restrict__ g2, const float* __restrict__ be2,
    const float* __restrict__ w3, const float* __restrict__ b3,
    float* __restrict__ featcat, int featoff,
    float* __restrict__ xnext, int Cin) {
  __shared__ bf16_t h[256 * 64];    // 32 KB  (GEMM1 A; later aliased)
  __shared__ bf16_t wl[32 * 64];    // 4 KB
  __shared__ bf16_t act1[256 * 32]; // 16 KB
  bf16_t* act2 = h;                 // alias: h dead after GEMM1
  bf16_t* act3 = h + 256 * 32;      // alias: upper half of h

  const int tid = threadIdx.x;
  const int f   = blockIdx.x >> 6;
  const int i0  = (blockIdx.x & 63) * 8;
  const int fi2 = 2 * Cin;

  { // build h rows: row = il*32 + kk ; [xi | xj - xi | 0-pad]
    int il = tid >> 5, kk = tid & 31;
    int i = i0 + il;
    int kuse = kk < 30 ? kk : 29;
    int j = idx[(f * 512 + i) * 32 + kuse];
    const float* xi = x + (f * 512 + i) * 32;
    const float* xj = x + (f * 512 + j) * 32;
    bf16_t* hr = h + tid * 64;
    for (int c = 0; c < Cin; ++c) {
      float a = xi[c], bb = xj[c];
      hr[c]        = (bf16_t)a;
      hr[Cin + c]  = (bf16_t)(bb - a);
    }
    for (int c = fi2; c < 64; ++c) hr[c] = (bf16_t)0.f;
  }
  load_w32(wl, w1, fi2);
  __syncthreads();
  edge_gemm(h, 64, (fi2 + 31) >> 5, wl, act1, b1, g1, be1);
  __syncthreads();
  load_w32(wl, w2, 32);
  __syncthreads();
  edge_gemm(act1, 32, 1, wl, act2, b2, g2, be2);
  __syncthreads();
  load_w32(wl, w3, 32);
  __syncthreads();
  edge_gemm(act2, 32, 1, wl, act3, b3, nullptr, nullptr);
  __syncthreads();
  { // max over the 30 real neighbors
    int il = tid >> 5, c = tid & 31;
    float m = -3.4e38f;
    for (int kk = 0; kk < 30; ++kk)
      m = fmaxf(m, (float)act3[(il * 32 + kk) * 32 + c]);
    int row = f * 512 + i0 + il;
    featcat[row * 96 + featoff + c] = m;
    xnext[row * 32 + c] = m;
  }
}

// ---------------- generic WMMA GEMM: C = A @ W^T + b ------------------------
// A:[M,K] f32, W:[N,K] f32. mode 0=store, 1=store+relu, 2=per-frame atomic max (frames of 512 rows)
__global__ void __launch_bounds__(256) k_gemm(
    const float* __restrict__ A, int lda,
    const float* __restrict__ W, int ldw,
    const float* __restrict__ bias,
    float* __restrict__ C, int ldc,
    int M, int N, int Ktot, int mode) {
  __shared__ bf16_t As[64 * 32];
  __shared__ bf16_t Bs[64 * 32];
  int nblocks = (N + 63) >> 6;
  int m0g = (blockIdx.x / nblocks) * 64;
  int n0g = (blockIdx.x % nblocks) * 64;
  int tid = threadIdx.x, wave = tid >> 5, lane = tid & 31;
  v8f acc[2];
  acc[0] = (v8f){0.f, 0.f, 0.f, 0.f, 0.f, 0.f, 0.f, 0.f};
  acc[1] = acc[0];
  int ktiles = (Ktot + 31) >> 5;
  for (int kt = 0; kt < ktiles; ++kt) {
    int k0 = kt * 32;
#pragma unroll
    for (int t = 0; t < 8; ++t) {
      int e = tid * 8 + t;
      int r = e >> 5, k = e & 31;
      int gr = m0g + r, gk = k0 + k;
      As[e] = (bf16_t)((gr < M && gk < Ktot) ? A[gr * lda + gk] : 0.f);
    }
#pragma unroll
    for (int t = 0; t < 8; ++t) {
      int e = tid * 8 + t;
      int n = e >> 5, k = e & 31;
      int gn = n0g + n, gk = k0 + k;
      Bs[e] = (bf16_t)((gn < N && gk < Ktot) ? W[gn * ldw + gk] : 0.f);
    }
    __syncthreads();
#pragma unroll
    for (int jj = 0; jj < 2; ++jj) {
      int job = wave + jj * 8;     // 4 m-tiles x 4 n-tiles
      v16bf a = load_a_frag(As, 32, (job >> 2) * 16, 0);
      v16bf b = load_b_frag(Bs, 32, (job & 3) * 16, 0);
      acc[jj] = wmma_bf16(a, b, acc[jj]);
    }
    __syncthreads();
  }
#pragma unroll
  for (int jj = 0; jj < 2; ++jj) {
    int job = wave + jj * 8;
    int mt = job >> 2, nt = job & 3;
    int col = n0g + nt * 16 + (lane & 15);
    int roff = (lane & 16) ? 8 : 0;
    if (col >= N) continue;
    float bb = bias ? bias[col] : 0.f;
    if (mode == 2) {
      float mx = -3.4e38f;
#pragma unroll
      for (int r = 0; r < 8; ++r) mx = fmaxf(mx, acc[jj][r] + bb);
      int grow = m0g + mt * 16 + roff;
      if (grow < M) atomicMaxF(&C[(grow >> 9) * ldc + col], mx);
    } else {
#pragma unroll
      for (int r = 0; r < 8; ++r) {
        int grow = m0g + mt * 16 + roff + r;
        if (grow < M) {
          float v = acc[jj][r] + bb;
          if (mode == 1) v = fmaxf(v, 0.f);
          C[grow * ldc + col] = v;
        }
      }
    }
  }
}

// ---------------- small kernels ---------------------------------------------
__global__ void k_fill(float* p, float v, int n) {
  int i = blockIdx.x * blockDim.x + threadIdx.x;
  if (i < n) p[i] = v;
}

__global__ void k_addpos(float* E, const float* __restrict__ pos) {
  int i = blockIdx.x * blockDim.x + threadIdx.x;
  if (i < 64 * 1024) {
    int f = i >> 10, d = i & 1023;
    E[i] += pos[(f & 15) * 1024 + d];
  }
}

__global__ void __launch_bounds__(256) k_attn(const float* __restrict__ qkv, float* __restrict__ o) {
  __shared__ float sc[16][17];
  int b = blockIdx.x >> 2, hh = blockIdx.x & 3;
  int tid = threadIdx.x;
  int qi = tid >> 4, kj = tid & 15;
  const float* qrow = qkv + (b * 16 + qi) * 3072 + hh * 256;
  const float* krow = qkv + (b * 16 + kj) * 3072 + 1024 + hh * 256;
  float s = 0.f;
  for (int c = 0; c < 256; ++c) s = fmaf(qrow[c], krow[c], s);
  sc[qi][kj] = s * 0.0625f; // 1/sqrt(256)
  __syncthreads();
  if (tid < 16) {
    float mx = -3.4e38f;
    for (int j = 0; j < 16; ++j) mx = fmaxf(mx, sc[tid][j]);
    float sum = 0.f;
    for (int j = 0; j < 16; ++j) { float e = expf(sc[tid][j] - mx); sc[tid][j] = e; sum += e; }
    float inv = 1.f / sum;
    for (int j = 0; j < 16; ++j) sc[tid][j] *= inv;
  }
  __syncthreads();
  for (int c = kj; c < 256; c += 16) {
    float acc = 0.f;
    for (int j = 0; j < 16; ++j)
      acc = fmaf(sc[qi][j], qkv[(b * 16 + j) * 3072 + 2048 + hh * 256 + c], acc);
    o[(b * 16 + qi) * 1024 + hh * 256 + c] = acc;
  }
}

__global__ void __launch_bounds__(256) k_add_ln(
    const float* __restrict__ xin, const float* __restrict__ res,
    const float* __restrict__ g, const float* __restrict__ b,
    float* __restrict__ out) {
  __shared__ float red[2][8];
  int row = blockIdx.x, tid = threadIdx.x;
  float s = 0.f, s2 = 0.f;
  for (int c = tid; c < 1024; c += 256) {
    float v = xin[row * 1024 + c] + res[row * 1024 + c];
    s += v; s2 += v * v;
  }
  for (int off = 16; off > 0; off >>= 1) {
    s  += __shfl_down(s, off);
    s2 += __shfl_down(s2, off);
  }
  int wave = tid >> 5, lane = tid & 31;
  if (lane == 0) { red[0][wave] = s; red[1][wave] = s2; }
  __syncthreads();
  if (tid == 0) {
    float a = 0.f, a2 = 0.f;
    for (int w = 0; w < 8; ++w) { a += red[0][w]; a2 += red[1][w]; }
    red[0][0] = a; red[1][0] = a2;
  }
  __syncthreads();
  float mean = red[0][0] * (1.f / 1024.f);
  float var  = red[1][0] * (1.f / 1024.f) - mean * mean;
  float inv  = rsqrtf(var + 1e-5f);
  for (int c = tid; c < 1024; c += 256) {
    float v = xin[row * 1024 + c] + res[row * 1024 + c];
    out[row * 1024 + c] = (v - mean) * inv * g[c] + b[c];
  }
}

__global__ void k_meanT(const float* __restrict__ E, float* __restrict__ feat) {
  int i = blockIdx.x * blockDim.x + threadIdx.x;
  if (i < 4 * 1024) {
    int b = i >> 10, d = i & 1023;
    float s = 0.f;
    for (int t = 0; t < 16; ++t) s += E[(b * 16 + t) * 1024 + d];
    feat[i] = s * (1.f / 16.f);
  }
}

__global__ void k_linear_small(const float* __restrict__ A, const float* __restrict__ W,
                               const float* __restrict__ bias, float* __restrict__ out,
                               int M, int N, int Kt, int relu) {
  int i = blockIdx.x * blockDim.x + threadIdx.x;
  if (i >= M * N) return;
  int m = i / N, n = i % N;
  float s = bias[n];
  for (int k = 0; k < Kt; ++k) s = fmaf(A[m * Kt + k], W[n * Kt + k], s);
  if (relu) s = fmaxf(s, 0.f);
  out[i] = s;
}

// ---------------- host launch -----------------------------------------------
extern "C" void kernel_launch(void* const* d_in, const int* in_sizes, int n_in,
                              void* d_out, int out_size, void* d_ws, size_t ws_size,
                              hipStream_t stream) {
  (void)in_sizes; (void)n_in; (void)out_size; (void)ws_size;
  const float* pc = (const float*)d_in[0];
  const float* fs = (const float*)d_in[1];
  auto P = [&](int i) { return (const float*)d_in[i]; };
  // JAX pytree (sorted-key) leaf order:
  // convs: per conv c (base=2+10c): L0 b,be,g,w ; L1 b,be,g,w ; L2 b,w
  // 32:ff1.b 33:ff1.w 34:ff2.b 35:ff2.w 36:in_proj.b 37:in_proj.w
  // 38:lin1.b 39:lin1.w 40:ln1.b 41:ln1.g 42:ln2.b 43:ln2.g
  // 44..51: out_mlp (b,w)x4  52:out_proj.b 53:out_proj.w 54:pos

  char* ws = (char*)d_ws;
  float* xA      = (float*)(ws);
  float* xB      = (float*)(ws + (4u  << 20));
  int*   idx     = (int*)  (ws + (8u  << 20));
  float* featcat = (float*)(ws + (12u << 20)); // 32768 x 96, exactly 12 MiB
  char*  b2      = ws + (24u << 20);
  float* E     = (float*)(b2);
  float* qkv   = (float*)(b2 + (256u  << 10));
  float* attno = (float*)(b2 + (1024u << 10));
  float* oproj = (float*)(b2 + (1280u << 10));
  float* E1    = (float*)(b2 + (1536u << 10));
  float* ffh   = (float*)(b2 + (1792u << 10));
  float* ff2o  = (float*)(b2 + (2304u << 10));
  float* E2    = (float*)(b2 + (2560u << 10));
  float* feat  = (float*)(b2 + (2816u << 10));
  float* h0 = feat + 4 * 1024;
  float* h1 = h0 + 4 * 1024;
  float* h2 = h1 + 4 * 256;

  k_build_x0<<<128, 256, 0, stream>>>(pc, fs, xA);
  // conv0 (Cin=15)
  k_knn<<<128, 256, 0, stream>>>(xA, idx);
  k_edge_conv<<<4096, 256, 0, stream>>>(xA, idx, P(5), P(2), P(4), P(3),
                                        P(9), P(6), P(8), P(7), P(11), P(10),
                                        featcat, 0, xB, 15);
  // conv1 (Cin=32)
  k_knn<<<128, 256, 0, stream>>>(xB, idx);
  k_edge_conv<<<4096, 256, 0, stream>>>(xB, idx, P(15), P(12), P(14), P(13),
                                        P(19), P(16), P(18), P(17), P(21), P(20),
                                        featcat, 32, xA, 32);
  // conv2 (Cin=32)
  k_knn<<<128, 256, 0, stream>>>(xA, idx);
  k_edge_conv<<<4096, 256, 0, stream>>>(xA, idx, P(25), P(22), P(24), P(23),
                                        P(29), P(26), P(28), P(27), P(31), P(30),
                                        featcat, 64, xB, 32);
  // lin1 + fused per-frame max pool (atomic) -> E[64,1024]
  k_fill<<<(64 * 1024 + 255) / 256, 256, 0, stream>>>(E, -3.4e38f, 64 * 1024);
  k_gemm<<<(32768 / 64) * (1024 / 64), 256, 0, stream>>>(featcat, 96, P(39), 96, P(38),
                                                         E, 1024, 32768, 1024, 96, 2);
  k_addpos<<<(64 * 1024 + 255) / 256, 256, 0, stream>>>(E, P(54));
  // transformer encoder layer (post-norm)
  k_gemm<<<(3072 / 64), 256, 0, stream>>>(E, 1024, P(37), 1024, P(36), qkv, 3072, 64, 3072, 1024, 0);
  k_attn<<<16, 256, 0, stream>>>(qkv, attno);
  k_gemm<<<(1024 / 64), 256, 0, stream>>>(attno, 1024, P(53), 1024, P(52), oproj, 1024, 64, 1024, 1024, 0);
  k_add_ln<<<64, 256, 0, stream>>>(E, oproj, P(41), P(40), E1);
  k_gemm<<<(2048 / 64), 256, 0, stream>>>(E1, 1024, P(33), 1024, P(32), ffh, 2048, 64, 2048, 1024, 1);
  k_gemm<<<(1024 / 64), 256, 0, stream>>>(ffh, 2048, P(35), 2048, P(34), ff2o, 1024, 64, 1024, 2048, 0);
  k_add_ln<<<64, 256, 0, stream>>>(E1, ff2o, P(43), P(42), E2);
  // head
  k_meanT<<<16, 256, 0, stream>>>(E2, feat);
  k_linear_small<<<(4 * 1024 + 255) / 256, 256, 0, stream>>>(feat, P(45), P(44), h0, 4, 1024, 1024, 1);
  k_linear_small<<<(4 * 256 + 255) / 256, 256, 0, stream>>>(h0, P(47), P(46), h1, 4, 256, 1024, 1);
  k_linear_small<<<(4 * 128 + 255) / 256, 256, 0, stream>>>(h1, P(49), P(48), h2, 4, 128, 256, 1);
  k_linear_small<<<(4 * 49 + 255) / 256, 256, 0, stream>>>(h2, P(51), P(50), (float*)d_out, 4, 49, 128, 0);
}
